// GNN_encode_80212809220209
// MI455X (gfx1250) — compile-verified, matrix operand order checked by reading, not run on hardware
//
#include <hip/hip_runtime.h>

// ---------------------------------------------------------------------------
// GNN encode + 2-step message passing for MI455X (gfx1250, wave32, WMMA).
// All 128-wide GEMMs on v_wmma_f32_16x16x32_bf16 (f32 accumulate).
// Intermediates staged in LDS as bf16; LayerNorm fully in registers via
// shfl_xor; weight K-slabs double-buffered (global fetch overlaps WMMA);
// bf16 mirrors of x and e so all A-fragment gathers are 16B vector loads.
// ---------------------------------------------------------------------------

typedef __bf16 bf16;
typedef __attribute__((ext_vector_type(16))) __bf16 bf16x16;
typedef __attribute__((ext_vector_type(8)))  __bf16 bf16x8;
typedef __attribute__((ext_vector_type(8)))  float  f32x8;

#define LNEPS 1e-5f
#define HST 136   // bf16 stage row stride: 272B rows -> 16B aligned, 4-bank skew

__device__ __forceinline__ f32x8 wmma_bf16(bf16x16 a, bf16x16 b, f32x8 c) {
  return __builtin_amdgcn_wmma_f32_16x16x32_bf16(false, a, false, b, (short)0, c,
                                                 false, false);
}

// A/B fragment, bf16 source. ISA 16-bit A layout: lane = M row (or N col for B),
// half = lane>>4 selects K octets {half*8, 16+half*8}.
__device__ __forceinline__ bf16x16 frag_ld_bf16(const bf16* row, int kbase, int half) {
  bf16x8 lo = *(const bf16x8*)(row + kbase + half * 8);
  bf16x8 hi = *(const bf16x8*)(row + kbase + 16 + half * 8);
  bf16x16 r;
#pragma unroll
  for (int j = 0; j < 8; ++j) { r[j] = lo[j]; r[8 + j] = hi[j]; }
  return r;
}

// Fragment from f32 source with scale (agg/cnt path).
__device__ __forceinline__ bf16x16 frag_ld_f32s(const float* row, int kbase, int half,
                                                float s) {
  bf16x16 r;
#pragma unroll
  for (int j = 0; j < 8; ++j) {
    r[j]     = (bf16)(row[kbase + half * 8 + j] * s);
    r[8 + j] = (bf16)(row[kbase + 16 + half * 8 + j] * s);
  }
  return r;
}

// Encoder input fragment: (v - mean) * rstd from LDS-cached params, K padded to 32.
__device__ __forceinline__ bf16x16 frag_input(const float* in, int row, int Din,
                                              const float* sMean, const float* sRstd,
                                              int half) {
  bf16x16 r;
#pragma unroll
  for (int j = 0; j < 8; ++j) {
    int k0 = half * 8 + j;
    int k1 = 16 + half * 8 + j;
    int c0 = k0 < Din ? k0 : Din - 1;
    int c1 = k1 < Din ? k1 : Din - 1;
    float v0 = (k0 < Din) ? (in[(size_t)row * Din + c0] - sMean[c0]) * sRstd[c0] : 0.f;
    float v1 = (k1 < Din) ? (in[(size_t)row * Din + c1] - sMean[c1]) * sRstd[c1] : 0.f;
    r[j] = (bf16)v0;
    r[8 + j] = (bf16)v1;
  }
  return r;
}

// B fragment from an LDS weight slab slab[n][k0..31] (n = output column).
__device__ __forceinline__ bf16x16 bfrag_from_slab(const bf16* slab, int nt, int lane) {
  int n = lane & 15, half = lane >> 4;
  return frag_ld_bf16(slab + (nt * 16 + n) * 32, 0, half);
}

// Double-buffered slab pipeline: fetch (global->regs), commit (regs->LDS).
struct SlabRegs { bf16x8 r[4]; };

__device__ __forceinline__ SlabRegs fetch_slab(const bf16* Wt, int Kpad, int kbase,
                                               int tid) {
  const bf16x8* src = (const bf16x8*)(Wt + (size_t)tid * Kpad + kbase);
  SlabRegs s;
  s.r[0] = src[0]; s.r[1] = src[1]; s.r[2] = src[2]; s.r[3] = src[3];
  return s;
}

__device__ __forceinline__ void commit_slab(bf16* slab, const SlabRegs& s, int tid) {
  bf16x8* dst = (bf16x8*)(slab + tid * 32);
  dst[0] = s.r[0]; dst[1] = s.r[1]; dst[2] = s.r[2]; dst[3] = s.r[3];
}

// Store C/D tile (bias + optional ELU) into the wave's bf16 stage.
// C layout: VGPR v -> row (v + 8*(lane>>4)), col = nt*16 + (lane&15).
__device__ __forceinline__ void store_h(bf16* st, int nt, int lane, f32x8 acc,
                                        const float* bias, bool do_elu) {
  int n = lane & 15, mh = lane >> 4;
  float b = bias[nt * 16 + n];
#pragma unroll
  for (int v = 0; v < 8; ++v) {
    float h = acc[v] + b;
    if (do_elu) h = h > 0.f ? h : __expf(h) - 1.f;
    st[(v + 8 * mh) * HST + nt * 16 + n] = (bf16)h;
  }
}

__device__ __forceinline__ void zero_acc(f32x8 acc[8]) {
#pragma unroll
  for (int q = 0; q < 8; ++q)
#pragma unroll
    for (int e = 0; e < 8; ++e) acc[q][e] = 0.f;
}

// 16-lane-group row reduction (rows of C layout live within one half-wave).
__device__ __forceinline__ float group_reduce(float v) {
  v += __shfl_xor(v, 1);
  v += __shfl_xor(v, 2);
  v += __shfl_xor(v, 4);
  v += __shfl_xor(v, 8);
  return v;
}

// ---------------------------------------------------------------------------
// Utility kernels
// ---------------------------------------------------------------------------
__global__ void zero_f32_kernel(float* p, long long n) {
  long long i = (long long)blockIdx.x * blockDim.x + threadIdx.x;
  long long stride = (long long)gridDim.x * blockDim.x;
  for (; i < n; i += stride) p[i] = 0.f;
}

__global__ void count_deg_kernel(const int* ei, float* cnt, int Ecount) {
  int i = blockIdx.x * blockDim.x + threadIdx.x;
  if (i < Ecount) atomicAdd(&cnt[ei[Ecount + i]], 1.f);
}

// Transpose + pad + cvt weights: W[k][n] f32 -> Wt[n][k] bf16 (k padded to Kpad).
__global__ void cvt_wt_kernel(const float* W, bf16* Wt, int K, int Kpad) {
  int i = blockIdx.x * blockDim.x + threadIdx.x;
  int total = 128 * Kpad;
  if (i < total) {
    int n = i / Kpad, k = i - n * Kpad;
    Wt[i] = (k < K) ? (bf16)W[(size_t)k * 128 + n] : (bf16)0.f;
  }
}

// ---------------------------------------------------------------------------
// Encoder: Lin(Din->128) ELU, Lin(128) ELU, Lin(128), LayerNorm.
// 4 waves/block, one 16-row tile per wave.
// ---------------------------------------------------------------------------
__global__ __launch_bounds__(128) void encode_kernel(
    const float* __restrict__ in, int Din, const float* __restrict__ mean,
    const float* __restrict__ stdv, const bf16* __restrict__ W0t,
    const float* __restrict__ b0, const bf16* __restrict__ W1t,
    const float* __restrict__ b1, const bf16* __restrict__ W2t,
    const float* __restrict__ b2, const float* __restrict__ g,
    const float* __restrict__ bt, float* __restrict__ out32,
    bf16* __restrict__ outbf, int R) {
  __shared__ bf16 slabs[2][128 * 32];
  __shared__ bf16 hstage[4][16 * HST];
  __shared__ float sMean[16], sRstd[16];

  int tid = threadIdx.x, wave = tid >> 5, lane = tid & 31;
  int m = lane & 15, half = lane >> 4, n = m, mh = half;
  bf16* hst = hstage[wave];

  if (tid < 16) {
    int k = tid < Din ? tid : Din - 1;
    sMean[tid] = mean[k];
    sRstd[tid] = 1.f / stdv[k];
  }
  __syncthreads();

  int tilesTotal = (R + 15) / 16;
  int tpi = gridDim.x * 4;
  int iters = (tilesTotal + tpi - 1) / tpi;

  for (int it = 0; it < iters; ++it) {
    int tile = (int)blockIdx.x * 4 + wave + it * tpi;
    bool act = tile < tilesTotal;
    int t0 = act ? tile : 0;
    int row = min(t0 * 16 + m, R - 1);

    f32x8 acc[8];
    zero_acc(acc);

    // ---- layer 0: K = 32 (Din padded) ----
    SlabRegs pre = fetch_slab(W0t, 32, 0, tid);
    commit_slab(slabs[0], pre, tid);
    __syncthreads();
    {
      bf16x16 a = frag_input(in, row, Din, sMean, sRstd, half);
#pragma unroll
      for (int nt = 0; nt < 8; ++nt)
        acc[nt] = wmma_bf16(a, bfrag_from_slab(slabs[0], nt, lane), acc[nt]);
    }
    __syncthreads();
#pragma unroll
    for (int nt = 0; nt < 8; ++nt) store_h(hst, nt, lane, acc[nt], b0, true);

    // ---- layer 1: 128x128, ELU ----
    zero_acc(acc);
    pre = fetch_slab(W1t, 128, 0, tid);
    commit_slab(slabs[0], pre, tid);
    __syncthreads();
    for (int kt = 0; kt < 4; ++kt) {
      const bf16* cur = slabs[kt & 1];
      if (kt < 3) pre = fetch_slab(W1t, 128, (kt + 1) * 32, tid);
      bf16x16 a = frag_ld_bf16(hst + m * HST, kt * 32, half);
#pragma unroll
      for (int nt = 0; nt < 8; ++nt)
        acc[nt] = wmma_bf16(a, bfrag_from_slab(cur, nt, lane), acc[nt]);
      if (kt < 3) commit_slab(slabs[(kt + 1) & 1], pre, tid);
      __syncthreads();
    }
#pragma unroll
    for (int nt = 0; nt < 8; ++nt) store_h(hst, nt, lane, acc[nt], b1, true);

    // ---- layer 2: 128x128, no activation ----
    zero_acc(acc);
    pre = fetch_slab(W2t, 128, 0, tid);
    commit_slab(slabs[0], pre, tid);
    __syncthreads();
    for (int kt = 0; kt < 4; ++kt) {
      const bf16* cur = slabs[kt & 1];
      if (kt < 3) pre = fetch_slab(W2t, 128, (kt + 1) * 32, tid);
      bf16x16 a = frag_ld_bf16(hst + m * HST, kt * 32, half);
#pragma unroll
      for (int nt = 0; nt < 8; ++nt)
        acc[nt] = wmma_bf16(a, bfrag_from_slab(cur, nt, lane), acc[nt]);
      if (kt < 3) commit_slab(slabs[(kt + 1) & 1], pre, tid);
      __syncthreads();
    }

    // ---- bias + register-resident LayerNorm ----
    float sum[8], sq[8];
#pragma unroll
    for (int v = 0; v < 8; ++v) { sum[v] = 0.f; sq[v] = 0.f; }
#pragma unroll
    for (int nt = 0; nt < 8; ++nt) {
      float b = b2[nt * 16 + n];
#pragma unroll
      for (int v = 0; v < 8; ++v) {
        float h = acc[nt][v] + b;
        acc[nt][v] = h;
        sum[v] += h;
        sq[v] += h * h;
      }
    }
#pragma unroll
    for (int v = 0; v < 8; ++v) {
      sum[v] = group_reduce(sum[v]);
      sq[v] = group_reduce(sq[v]);
    }
    float mu[8], rs[8];
#pragma unroll
    for (int v = 0; v < 8; ++v) {
      mu[v] = sum[v] * (1.f / 128.f);
      float var = sq[v] * (1.f / 128.f) - mu[v] * mu[v];
      rs[v] = rsqrtf(var + LNEPS);
    }
#pragma unroll
    for (int nt = 0; nt < 8; ++nt) {
      int col = nt * 16 + n;
      float gg = g[col], bb = bt[col];
#pragma unroll
      for (int v = 0; v < 8; ++v) {
        int rl = v + 8 * mh;
        int gr = tile * 16 + rl;
        float o = (acc[nt][v] - mu[v]) * rs[v] * gg + bb;
        if (act && gr < R) {
          out32[(size_t)gr * 128 + col] = o;
          if (outbf) outbf[(size_t)gr * 128 + col] = (bf16)o;
        }
      }
    }
  }
}

// ---------------------------------------------------------------------------
// Edge update: e' = LN(e + MLP(cat[x_src, x_dst, e])); scatter-add e' to agg[dst].
// All A-fragment gathers are bf16 16B vector loads (xbf, ebf); f32 copy of e
// is used only for the residual/LN path.
// ---------------------------------------------------------------------------
__global__ __launch_bounds__(128) void edge_update_kernel(
    const bf16* __restrict__ xbf, float* __restrict__ e32,
    bf16* __restrict__ ebf, const int* __restrict__ ei,
    const bf16* __restrict__ W0t, const float* __restrict__ b0,
    const bf16* __restrict__ W1t, const float* __restrict__ b1,
    const float* __restrict__ g, const float* __restrict__ bt,
    float* __restrict__ agg, int Ecount) {
  __shared__ bf16 slabs[2][128 * 32];
  __shared__ bf16 hstage[4][16 * HST];

  int tid = threadIdx.x, wave = tid >> 5, lane = tid & 31;
  int m = lane & 15, half = lane >> 4, n = m, mh = half;
  bf16* hst = hstage[wave];

  int tilesTotal = (Ecount + 15) / 16;
  int tpi = gridDim.x * 4;
  int iters = (tilesTotal + tpi - 1) / tpi;

  for (int it = 0; it < iters; ++it) {
    int tile = (int)blockIdx.x * 4 + wave + it * tpi;
    bool act = tile < tilesTotal;
    int t0 = act ? tile : 0;
    int eId = min(t0 * 16 + m, Ecount - 1);
    int sRow = ei[eId];
    int dRow = ei[Ecount + eId];

    // Warm caches for the latency-critical gathers (global_prefetch_b8).
    {
      const char* ps = (const char*)(xbf + (size_t)sRow * 128);
      const char* pd = (const char*)(xbf + (size_t)dRow * 128);
      const char* pb = (const char*)(ebf + (size_t)eId * 128);
      const char* pe = (const char*)(e32 + (size_t)eId * 128);
      __builtin_prefetch(ps, 0, 1);
      __builtin_prefetch(ps + 128, 0, 1);
      __builtin_prefetch(pd, 0, 1);
      __builtin_prefetch(pd + 128, 0, 1);
      __builtin_prefetch(pb, 0, 1);
      __builtin_prefetch(pe, 0, 1);
      __builtin_prefetch(pe + 256, 0, 1);
    }

    f32x8 acc[8];
    zero_acc(acc);

    // ---- layer 0: K = 384 = [x_src | x_dst | e] ----
    SlabRegs pre = fetch_slab(W0t, 384, 0, tid);
    commit_slab(slabs[0], pre, tid);
    __syncthreads();
    for (int kt = 0; kt < 12; ++kt) {
      const bf16* cur = slabs[kt & 1];
      if (kt < 11) pre = fetch_slab(W0t, 384, (kt + 1) * 32, tid);
      bf16x16 a;
      if (kt < 4)
        a = frag_ld_bf16(xbf + (size_t)sRow * 128, kt * 32, half);
      else if (kt < 8)
        a = frag_ld_bf16(xbf + (size_t)dRow * 128, (kt - 4) * 32, half);
      else
        a = frag_ld_bf16(ebf + (size_t)eId * 128, (kt - 8) * 32, half);
#pragma unroll
      for (int nt = 0; nt < 8; ++nt)
        acc[nt] = wmma_bf16(a, bfrag_from_slab(cur, nt, lane), acc[nt]);
      if (kt < 11) commit_slab(slabs[(kt + 1) & 1], pre, tid);
      __syncthreads();
    }
#pragma unroll
    for (int nt = 0; nt < 8; ++nt) store_h(hst, nt, lane, acc[nt], b0, true);

    // ---- layer 1: 128x128 ----
    zero_acc(acc);
    pre = fetch_slab(W1t, 128, 0, tid);
    commit_slab(slabs[0], pre, tid);
    __syncthreads();
    for (int kt = 0; kt < 4; ++kt) {
      const bf16* cur = slabs[kt & 1];
      if (kt < 3) pre = fetch_slab(W1t, 128, (kt + 1) * 32, tid);
      bf16x16 a = frag_ld_bf16(hst + m * HST, kt * 32, half);
#pragma unroll
      for (int nt = 0; nt < 8; ++nt)
        acc[nt] = wmma_bf16(a, bfrag_from_slab(cur, nt, lane), acc[nt]);
      if (kt < 3) commit_slab(slabs[(kt + 1) & 1], pre, tid);
      __syncthreads();
    }

    // ---- bias + residual + register LayerNorm + scatter ----
    float sum[8], sq[8];
#pragma unroll
    for (int v = 0; v < 8; ++v) { sum[v] = 0.f; sq[v] = 0.f; }
#pragma unroll
    for (int nt = 0; nt < 8; ++nt) {
      float b = b1[nt * 16 + n];
#pragma unroll
      for (int v = 0; v < 8; ++v) {
        int rl = v + 8 * mh;
        int eS = __shfl(eId, rl);
        float r = e32[(size_t)eS * 128 + nt * 16 + n];
        float h = acc[nt][v] + b + r;
        acc[nt][v] = h;
        sum[v] += h;
        sq[v] += h * h;
      }
    }
#pragma unroll
    for (int v = 0; v < 8; ++v) {
      sum[v] = group_reduce(sum[v]);
      sq[v] = group_reduce(sq[v]);
    }
    float mu[8], rs[8];
#pragma unroll
    for (int v = 0; v < 8; ++v) {
      mu[v] = sum[v] * (1.f / 128.f);
      float var = sq[v] * (1.f / 128.f) - mu[v] * mu[v];
      rs[v] = rsqrtf(var + LNEPS);
    }
#pragma unroll
    for (int nt = 0; nt < 8; ++nt) {
      int col = nt * 16 + n;
      float gg = g[col], bb = bt[col];
#pragma unroll
      for (int v = 0; v < 8; ++v) {
        int rl = v + 8 * mh;
        int ge = tile * 16 + rl;
        int eS = __shfl(eId, rl);
        int dS = __shfl(dRow, rl);
        float o = (acc[nt][v] - mu[v]) * rs[v] * gg + bb;
        if (act && ge < Ecount) {
          e32[(size_t)eS * 128 + col] = o;
          ebf[(size_t)eS * 128 + col] = (bf16)o;
          atomicAdd(&agg[(size_t)dS * 128 + col], o);  // segment-sum scatter
        }
      }
    }
  }
}

// ---------------------------------------------------------------------------
// Node update: x' = LN(x + MLP(cat[x, agg/cnt])).
// ---------------------------------------------------------------------------
__global__ __launch_bounds__(128) void node_update_kernel(
    const float* __restrict__ x32in, bf16* __restrict__ xbf,
    const float* __restrict__ agg, const float* __restrict__ cnt,
    const bf16* __restrict__ W0t, const float* __restrict__ b0,
    const bf16* __restrict__ W1t, const float* __restrict__ b1,
    const float* __restrict__ g, const float* __restrict__ bt,
    float* __restrict__ x32out, int Nn) {
  __shared__ bf16 slabs[2][128 * 32];
  __shared__ bf16 hstage[4][16 * HST];

  int tid = threadIdx.x, wave = tid >> 5, lane = tid & 31;
  int m = lane & 15, half = lane >> 4, n = m, mh = half;
  bf16* hst = hstage[wave];

  int tilesTotal = (Nn + 15) / 16;
  int tpi = gridDim.x * 4;
  int iters = (tilesTotal + tpi - 1) / tpi;

  for (int it = 0; it < iters; ++it) {
    int tile = (int)blockIdx.x * 4 + wave + it * tpi;
    bool act = tile < tilesTotal;
    int t0 = act ? tile : 0;
    int row = min(t0 * 16 + m, Nn - 1);
    float sc = 1.f / fmaxf(cnt[row], 1.f);  // scatter-mean divisor

    f32x8 acc[8];
    zero_acc(acc);

    // ---- layer 0: K = 256 = [x | agg/cnt] ----
    SlabRegs pre = fetch_slab(W0t, 256, 0, tid);
    commit_slab(slabs[0], pre, tid);
    __syncthreads();
    for (int kt = 0; kt < 8; ++kt) {
      const bf16* cur = slabs[kt & 1];
      if (kt < 7) pre = fetch_slab(W0t, 256, (kt + 1) * 32, tid);
      bf16x16 a;
      if (kt < 4)
        a = frag_ld_bf16(xbf + (size_t)row * 128, kt * 32, half);
      else
        a = frag_ld_f32s(agg + (size_t)row * 128, (kt - 4) * 32, half, sc);
#pragma unroll
      for (int nt = 0; nt < 8; ++nt)
        acc[nt] = wmma_bf16(a, bfrag_from_slab(cur, nt, lane), acc[nt]);
      if (kt < 7) commit_slab(slabs[(kt + 1) & 1], pre, tid);
      __syncthreads();
    }
#pragma unroll
    for (int nt = 0; nt < 8; ++nt) store_h(hst, nt, lane, acc[nt], b0, true);

    // ---- layer 1: 128x128 ----
    zero_acc(acc);
    pre = fetch_slab(W1t, 128, 0, tid);
    commit_slab(slabs[0], pre, tid);
    __syncthreads();
    for (int kt = 0; kt < 4; ++kt) {
      const bf16* cur = slabs[kt & 1];
      if (kt < 3) pre = fetch_slab(W1t, 128, (kt + 1) * 32, tid);
      bf16x16 a = frag_ld_bf16(hst + m * HST, kt * 32, half);
#pragma unroll
      for (int nt = 0; nt < 8; ++nt)
        acc[nt] = wmma_bf16(a, bfrag_from_slab(cur, nt, lane), acc[nt]);
      if (kt < 3) commit_slab(slabs[(kt + 1) & 1], pre, tid);
      __syncthreads();
    }

    // ---- bias + residual + register LayerNorm ----
    float sum[8], sq[8];
#pragma unroll
    for (int v = 0; v < 8; ++v) { sum[v] = 0.f; sq[v] = 0.f; }
#pragma unroll
    for (int nt = 0; nt < 8; ++nt) {
      float b = b1[nt * 16 + n];
#pragma unroll
      for (int v = 0; v < 8; ++v) {
        int rl = v + 8 * mh;
        int rS = __shfl(row, rl);
        float r = x32in[(size_t)rS * 128 + nt * 16 + n];
        float h = acc[nt][v] + b + r;
        acc[nt][v] = h;
        sum[v] += h;
        sq[v] += h * h;
      }
    }
#pragma unroll
    for (int v = 0; v < 8; ++v) {
      sum[v] = group_reduce(sum[v]);
      sq[v] = group_reduce(sq[v]);
    }
    float mu[8], rs[8];
#pragma unroll
    for (int v = 0; v < 8; ++v) {
      mu[v] = sum[v] * (1.f / 128.f);
      float var = sq[v] * (1.f / 128.f) - mu[v] * mu[v];
      rs[v] = rsqrtf(var + LNEPS);
    }
#pragma unroll
    for (int nt = 0; nt < 8; ++nt) {
      int col = nt * 16 + n;
      float gg = g[col], bb = bt[col];
#pragma unroll
      for (int v = 0; v < 8; ++v) {
        int rl = v + 8 * mh;
        int gr = tile * 16 + rl;
        float o = (acc[nt][v] - mu[v]) * rs[v] * gg + bb;
        if (act && gr < Nn) {
          x32out[(size_t)gr * 128 + col] = o;
          xbf[(size_t)gr * 128 + col] = (bf16)o;
        }
      }
    }
  }
}

// ---------------------------------------------------------------------------
// Host orchestration
// ---------------------------------------------------------------------------
extern "C" void kernel_launch(void* const* d_in, const int* in_sizes, int n_in,
                              void* d_out, int out_size, void* d_ws, size_t ws_size,
                              hipStream_t stream) {
  (void)in_sizes; (void)n_in; (void)out_size; (void)ws_size;
  const int N = 50000, E = 800000, H = 128;

  const float* x      = (const float*)d_in[0];
  const int*   ei     = (const int*)d_in[1];
  const float* ea     = (const float*)d_in[2];
  const float* mean_x = (const float*)d_in[3];
  const float* std_x  = (const float*)d_in[4];
  const float* mean_e = (const float*)d_in[5];
  const float* std_e  = (const float*)d_in[6];
  const float* ne_W0  = (const float*)d_in[7];
  const float* ne_b0  = (const float*)d_in[8];
  const float* ne_W   = (const float*)d_in[9];
  const float* ne_b   = (const float*)d_in[10];
  const float* ne_g   = (const float*)d_in[11];
  const float* ne_bt  = (const float*)d_in[12];
  const float* ee_W0  = (const float*)d_in[13];
  const float* ee_b0  = (const float*)d_in[14];
  const float* ee_W   = (const float*)d_in[15];
  const float* ee_b   = (const float*)d_in[16];
  const float* ee_g   = (const float*)d_in[17];
  const float* ee_bt  = (const float*)d_in[18];
  const float* emp_W0 = (const float*)d_in[19];
  const float* emp_b0 = (const float*)d_in[20];
  const float* emp_W1 = (const float*)d_in[21];
  const float* emp_b1 = (const float*)d_in[22];
  const float* emp_g  = (const float*)d_in[23];
  const float* emp_bt = (const float*)d_in[24];
  const float* nmp_W0 = (const float*)d_in[25];
  const float* nmp_b0 = (const float*)d_in[26];
  const float* nmp_W1 = (const float*)d_in[27];
  const float* nmp_b1 = (const float*)d_in[28];
  const float* nmp_g  = (const float*)d_in[29];
  const float* nmp_bt = (const float*)d_in[30];

  char* ws = (char*)d_ws;
  size_t off = 0;
  auto ALLOC = [&](size_t bytes) -> char* {
    char* p = ws + off;
    off = (off + bytes + 255) & ~(size_t)255;
    return p;
  };

  float* x32 = (float*)ALLOC((size_t)N * H * 4);
  bf16*  xbf = (bf16*) ALLOC((size_t)N * H * 2);
  float* e32 = (float*)ALLOC((size_t)E * H * 4);
  bf16*  ebf = (bf16*) ALLOC((size_t)E * H * 2);
  float* agg = (float*)ALLOC((size_t)N * H * 4);
  float* cnt = (float*)ALLOC((size_t)N * 4);

  bf16* neW0t = (bf16*)ALLOC(128 * 32 * 2);
  bf16* neW1t = (bf16*)ALLOC(128 * 128 * 2);
  bf16* neW2t = (bf16*)ALLOC(128 * 128 * 2);
  bf16* eeW0t = (bf16*)ALLOC(128 * 32 * 2);
  bf16* eeW1t = (bf16*)ALLOC(128 * 128 * 2);
  bf16* eeW2t = (bf16*)ALLOC(128 * 128 * 2);
  bf16* empW0t[2]; bf16* empW1t[2]; bf16* nmpW0t[2]; bf16* nmpW1t[2];
  for (int i = 0; i < 2; ++i) {
    empW0t[i] = (bf16*)ALLOC(128 * 384 * 2);
    empW1t[i] = (bf16*)ALLOC(128 * 128 * 2);
    nmpW0t[i] = (bf16*)ALLOC(128 * 256 * 2);
    nmpW1t[i] = (bf16*)ALLOC(128 * 128 * 2);
  }

  auto cvt = [&](const float* W, bf16* Wt, int K, int Kpad) {
    int total = 128 * Kpad;
    cvt_wt_kernel<<<(total + 127) / 128, 128, 0, stream>>>(W, Wt, K, Kpad);
  };

  // ---- weight prep (transposed bf16, K padded to tile multiples) ----
  cvt(ne_W0, neW0t, 11, 32);
  cvt(ne_W, neW1t, 128, 128);
  cvt(ne_W + 128 * 128, neW2t, 128, 128);
  cvt(ee_W0, eeW0t, 3, 32);
  cvt(ee_W, eeW1t, 128, 128);
  cvt(ee_W + 128 * 128, eeW2t, 128, 128);
  for (int i = 0; i < 2; ++i) {
    cvt(emp_W0 + (size_t)i * 384 * 128, empW0t[i], 384, 384);
    cvt(emp_W1 + (size_t)i * 128 * 128, empW1t[i], 128, 128);
    cvt(nmp_W0 + (size_t)i * 256 * 128, nmpW0t[i], 256, 256);
    cvt(nmp_W1 + (size_t)i * 128 * 128, nmpW1t[i], 128, 128);
  }

  // ---- degrees ----
  zero_f32_kernel<<<1024, 256, 0, stream>>>(cnt, (long long)N);
  count_deg_kernel<<<(E + 255) / 256, 256, 0, stream>>>(ei, cnt, E);

  // ---- encoders ----
  int gridN = ((N + 15) / 16 + 3) / 4;  // 4 wave-tiles / block
  int gridE = ((E + 15) / 16 + 3) / 4;
  encode_kernel<<<gridN, 128, 0, stream>>>(x, 11, mean_x, std_x, neW0t, ne_b0,
                                           neW1t, ne_b, neW2t, ne_b + 128, ne_g,
                                           ne_bt, x32, xbf, N);
  encode_kernel<<<gridE, 128, 0, stream>>>(ea, 3, mean_e, std_e, eeW0t, ee_b0,
                                           eeW1t, ee_b, eeW2t, ee_b + 128, ee_g,
                                           ee_bt, e32, ebf, E);

  // ---- message passing ----
  for (int i = 0; i < 2; ++i) {
    zero_f32_kernel<<<2048, 256, 0, stream>>>(agg, (long long)N * H);
    edge_update_kernel<<<gridE, 128, 0, stream>>>(
        xbf, e32, ebf, ei, empW0t[i], emp_b0 + i * 128, empW1t[i],
        emp_b1 + i * 128, emp_g + i * 128, emp_bt + i * 128, agg, E);
    float* xout = (i == 1) ? (float*)d_out : x32;
    node_update_kernel<<<gridN, 128, 0, stream>>>(
        x32, xbf, agg, cnt, nmpW0t[i], nmp_b0 + i * 128, nmpW1t[i],
        nmp_b1 + i * 128, nmp_g + i * 128, nmp_bt + i * 128, xout, N);
  }
}
